// client_HGPSLPool_7997229105404
// MI455X (gfx1250) — matrix operand order repaired and checked by reading, not compiled, not run on hardware
//
#include <hip/hip_runtime.h>
#include <hip/hip_bf16.h>

// ---------------------------------------------------------------------------
// HGPSL forward for MI455X (gfx1250, wave32, WMMA).
// GEMMs: fp32 masters converted once to f16 (A row-major, B pre-transposed),
// then V_WMMA_F32_16X16X32_F16 with f32 accumulation and vector b128 loads.
// ---------------------------------------------------------------------------

typedef __attribute__((ext_vector_type(16))) _Float16 v16h;
typedef __attribute__((ext_vector_type(8)))  _Float16 v8h;
typedef __attribute__((ext_vector_type(8)))  float    v8f;

static constexpr int CB  = 64;   // batch
static constexpr int CN  = 512;  // nodes
static constexpr int CH  = 128;  // hidden / F_IN
static constexpr int CK1 = 256;  // nodes after pool1
static constexpr int CK2 = 128;  // nodes after pool2
static constexpr int CC  = 10;   // classes
static constexpr int CCP = 16;   // padded classes (Nc multiple of 16)
static constexpr float NEG_SLOPE = 0.2f;
static constexpr float LAMB      = 1.0f;
static constexpr float BIG_NEG   = -1e30f;

// ---------------------------------------------------------------------------
// f16 WMMA GEMM: C[b] (MxNc, f32) = A[b] (MxK f16 row-major)
//                                 x Bt[b] (NcxK f16 row-major == B^T) (+bias,relu)
// Requirements: M%16==0, Nc%16==0, K%32==0 (all call sites satisfy this).
// 4 waves / block, one 16x16 C tile per wave.
// A fragment: two 16B loads (K runs [16h,16h+8) and [16h+8,16h+16)).
// B fragment: one 32B load  (K run  [16h,16h+16)) from the transposed copy.
// ---------------------------------------------------------------------------
__global__ void k_wmma_gemm16(const _Float16* __restrict__ A,
                              const _Float16* __restrict__ Bt,
                              const float* __restrict__ bias,
                              float* __restrict__ C,
                              int M, int Nc, int K,
                              long long sA, long long sB, long long sC, int relu)
{
    const int tilesN = Nc >> 4;
    const int tilesM = M >> 4;
    const int tile   = blockIdx.x * (blockDim.x >> 5) + (threadIdx.x >> 5);
    if (tile >= tilesM * tilesN) return;
    const int tm   = (tile / tilesN) << 4;
    const int tn   = (tile % tilesN) << 4;
    const int lane = threadIdx.x & 31;
    const int half = lane >> 4;
    const int l16  = lane & 15;
    const int batch = blockIdx.y;

    const _Float16* Ap = A  + batch * sA + (long long)(tm + l16) * K + 16 * half;
    const _Float16* Bp = Bt + batch * sB + (long long)(tn + l16) * K + 16 * half;
    float* Cb = C + batch * sC;

    v8f acc = {};
    #pragma unroll 2
    for (int k0 = 0; k0 < K; k0 += 32) {
        v8h alo = *(const v8h*)(Ap + k0);
        v8h ahi = *(const v8h*)(Ap + k0 + 8);
        v16h af = __builtin_shufflevector(alo, ahi,
                    0, 1, 2, 3, 4, 5, 6, 7, 8, 9, 10, 11, 12, 13, 14, 15);
        v16h bf = *(const v16h*)(Bp + k0);
        acc = __builtin_amdgcn_wmma_f32_16x16x32_f16(
            false, af, false, bf, (short)0, acc, false, false);
    }

    #pragma unroll
    for (int r = 0; r < 8; ++r) {
        int row = tm + r + 8 * half;
        int col = tn + l16;
        float v = acc[r];
        if (bias) v += bias[col];
        if (relu) v = fmaxf(v, 0.0f);
        Cb[(long long)row * Nc + col] = v;
    }
}

// ---------------------------------------------------------------------------
// fp32 -> f16 convert (8 elements / thread, vector load+store).
// ---------------------------------------------------------------------------
__global__ void k_cvt8(const float* __restrict__ src, _Float16* __restrict__ dst,
                       long long tot8)
{
    long long i = (long long)blockIdx.x * blockDim.x + threadIdx.x;
    if (i >= tot8) return;
    const float4* s = (const float4*)src;
    float4 a = s[2 * i];
    float4 b = s[2 * i + 1];
    v8h o = { (_Float16)a.x, (_Float16)a.y, (_Float16)a.z, (_Float16)a.w,
              (_Float16)b.x, (_Float16)b.y, (_Float16)b.z, (_Float16)b.w };
    *(v8h*)(dst + 8 * i) = o;
}

// fp32 [batch][rows][cols] -> f16 transposed [batch][cols][rows]
__global__ void k_cvt_t(const float* __restrict__ src, _Float16* __restrict__ dst,
                        int rows, int cols, long long tot)
{
    long long idx = (long long)blockIdx.x * blockDim.x + threadIdx.x;
    if (idx >= tot) return;
    int rr = (int)(idx % rows);
    long long t = idx / rows;
    int cc = (int)(t % cols);
    long long b = t / cols;
    dst[idx] = (_Float16)src[(b * rows + rr) * cols + cc];
}

// Pad Wl3 (64 x 10) -> transposed f16 (16 x 64), bl3 (10) -> padded f32 (16).
__global__ void k_pad_w3(const float* __restrict__ W, const float* __restrict__ b,
                         _Float16* __restrict__ Wt, float* __restrict__ bp)
{
    int t = blockIdx.x * blockDim.x + threadIdx.x;
    if (t < CCP * 64) {
        int c = t / 64, k = t % 64;
        Wt[t] = (c < CC) ? (_Float16)W[(long long)k * CC + c] : (_Float16)0.0f;
    }
    if (t < CCP) bp[t] = (t < CC) ? b[t] : 0.0f;
}

// ---------------------------------------------------------------------------
// GCN normalization: deg of adj with diag replaced (0 -> 1), dinv = rsqrt.
// ---------------------------------------------------------------------------
__global__ void k_gcn_dinv(const float* __restrict__ adj, float* __restrict__ dinv, int n)
{
    int b = blockIdx.y;
    int i = blockIdx.x * blockDim.x + threadIdx.x;
    if (i >= n) return;
    const float* row = adj + ((long long)b * n + i) * n;
    float diag = row[i];
    float dsl  = (diag == 0.0f) ? 1.0f : diag;
    float s = dsl;
    for (int j = 0; j < n; ++j)
        if (j != i) s += row[j];
    dinv[b * n + i] = (s > 0.0f) ? rsqrtf(s) : 0.0f;
}

__global__ void k_gcn_norm(const float* __restrict__ adj, const float* __restrict__ dinv,
                           float* __restrict__ an, int n)
{
    long long idx = (long long)blockIdx.x * blockDim.x + threadIdx.x;
    long long tot = (long long)CB * n * n;
    if (idx >= tot) return;
    int j = (int)(idx % n);
    long long t = idx / n;
    int i = (int)(t % n);
    int b = (int)(t / n);
    float v;
    if (i == j) {
        float d = adj[((long long)b * n + i) * n + i];
        v = (d == 0.0f) ? 1.0f : d;
    } else {
        v = adj[idx];
    }
    an[idx] = dinv[b * n + i] * v * dinv[b * n + j];
}

// Plain symmetric norm for NodeInformationScore (no diag fill).
__global__ void k_row_dinv(const float* __restrict__ adj, float* __restrict__ dinv, int n)
{
    int b = blockIdx.y;
    int i = blockIdx.x * blockDim.x + threadIdx.x;
    if (i >= n) return;
    const float* row = adj + ((long long)b * n + i) * n;
    float s = 0.0f;
    for (int j = 0; j < n; ++j) s += row[j];
    dinv[b * n + i] = (s > 0.0f) ? rsqrtf(s) : 0.0f;
}

__global__ void k_sym_norm(const float* __restrict__ adj, const float* __restrict__ dinv,
                           float* __restrict__ an, int n)
{
    long long idx = (long long)blockIdx.x * blockDim.x + threadIdx.x;
    long long tot = (long long)CB * n * n;
    if (idx >= tot) return;
    int j = (int)(idx % n);
    long long t = idx / n;
    int i = (int)(t % n);
    int b = (int)(t / n);
    an[idx] = dinv[b * n + i] * adj[idx] * dinv[b * n + j];
}

__global__ void k_abs_diff_rowsum(const float* __restrict__ x, const float* __restrict__ y,
                                  float* __restrict__ score, int n, int f)
{
    int b = blockIdx.y;
    int i = blockIdx.x * blockDim.x + threadIdx.x;
    if (i >= n) return;
    const float* xr = x + ((long long)b * n + i) * f;
    const float* yr = y + ((long long)b * n + i) * f;
    float s = 0.0f;
    for (int c = 0; c < f; ++c) s += fabsf(xr[c] - yr[c]);
    score[b * n + i] = s;
}

// ---------------------------------------------------------------------------
// Top-k per batch (descending, ties -> lower index), LDS argmax iteration.
// blockDim.x == n, dyn LDS = 3*n*4 bytes.
// ---------------------------------------------------------------------------
__global__ void k_topk(const float* __restrict__ score, int* __restrict__ perm, int n, int k)
{
    extern __shared__ float sh[];
    float* vals = sh;
    float* rv   = sh + n;
    int*   ri   = (int*)(sh + 2 * n);
    int b = blockIdx.x;
    int t = threadIdx.x;
    vals[t] = score[(long long)b * n + t];
    __syncthreads();
    for (int sel = 0; sel < k; ++sel) {
        rv[t] = vals[t];
        ri[t] = t;
        __syncthreads();
        for (int s = n >> 1; s > 0; s >>= 1) {
            if (t < s) {
                float v2 = rv[t + s]; int i2 = ri[t + s];
                if (v2 > rv[t] || (v2 == rv[t] && i2 < ri[t])) { rv[t] = v2; ri[t] = i2; }
            }
            __syncthreads();
        }
        if (t == 0) {
            perm[(long long)b * k + sel] = ri[0];
            vals[ri[0]] = BIG_NEG;
        }
        __syncthreads();
    }
}

// ---------------------------------------------------------------------------
// Two-hop helpers.
// ---------------------------------------------------------------------------
__global__ void k_binarize(const float* __restrict__ a, float* __restrict__ s, long long tot)
{
    long long idx = (long long)blockIdx.x * blockDim.x + threadIdx.x;
    if (idx >= tot) return;
    s[idx] = (a[idx] > 0.0f) ? 1.0f : 0.0f;
}

__global__ void k_threshold_combine(float* __restrict__ s, const float* __restrict__ sq, long long tot)
{
    long long idx = (long long)blockIdx.x * blockDim.x + threadIdx.x;
    if (idx >= tot) return;
    s[idx] = ((s[idx] + sq[idx]) > 0.0f) ? 1.0f : 0.0f;
}

// ---------------------------------------------------------------------------
// Gather / sub-matrix with permutation.
// ---------------------------------------------------------------------------
__global__ void k_gather_rows(const float* __restrict__ src, const int* __restrict__ perm,
                              float* __restrict__ dst, int n, int k, int f)
{
    long long idx = (long long)blockIdx.x * blockDim.x + threadIdx.x;
    long long tot = (long long)CB * k * f;
    if (idx >= tot) return;
    int c = (int)(idx % f);
    long long t = idx / f;
    int i = (int)(t % k);
    int b = (int)(t / k);
    int row = perm[(long long)b * k + i];
    dst[idx] = src[((long long)b * n + row) * f + c];
}

__global__ void k_sub_matrix(const float* __restrict__ src, const int* __restrict__ perm,
                             float* __restrict__ dst, int n, int k, int add_eye_max)
{
    long long idx = (long long)blockIdx.x * blockDim.x + threadIdx.x;
    long long tot = (long long)CB * k * k;
    if (idx >= tot) return;
    int j = (int)(idx % k);
    long long t = idx / k;
    int i = (int)(t % k);
    int b = (int)(t / k);
    int pi = perm[(long long)b * k + i];
    int pj = perm[(long long)b * k + j];
    float v = src[((long long)b * n + pi) * n + pj];
    if (add_eye_max && i == j) v = fmaxf(v, 1.0f);
    dst[idx] = v;
}

// ---------------------------------------------------------------------------
// Attention logits for structure learning.
// ---------------------------------------------------------------------------
__global__ void k_att_dots(const float* __restrict__ xp, const float* __restrict__ att,
                           float* __restrict__ ai, float* __restrict__ aj, int k, int h)
{
    int b = blockIdx.y;
    int i = blockIdx.x * blockDim.x + threadIdx.x;
    if (i >= k) return;
    const float* row = xp + ((long long)b * k + i) * h;
    float s0 = 0.0f, s1 = 0.0f;
    for (int f = 0; f < h; ++f) {
        float v = row[f];
        s0 += v * att[f];
        s1 += v * att[h + f];
    }
    ai[b * k + i] = s0;
    aj[b * k + i] = s1;
}

__global__ void k_logits(const float* __restrict__ ai, const float* __restrict__ aj,
                         const float* __restrict__ ap, float* __restrict__ lg, int k)
{
    long long idx = (long long)blockIdx.x * blockDim.x + threadIdx.x;
    long long tot = (long long)CB * k * k;
    if (idx >= tot) return;
    int j = (int)(idx % k);
    long long t = idx / k;
    float v = ai[t] + aj[(t / k) * k + j];
    v = (v > 0.0f) ? v : NEG_SLOPE * v;
    lg[idx] = v + LAMB * ap[idx];
}

// ---------------------------------------------------------------------------
// Masked row sparsemax; block per row, blockDim == k (power of two),
// bitonic sort + Hillis-Steele scan in LDS. dyn LDS = 3*k*4 bytes.
// ---------------------------------------------------------------------------
__global__ void k_sparsemax(const float* __restrict__ lg, const float* __restrict__ sp,
                            float* __restrict__ out, int k)
{
    extern __shared__ float sh[];
    float* d   = sh;
    float* cum = sh + k;
    float* red = sh + 2 * k;
    long long row = blockIdx.x;
    int t = threadIdx.x;
    long long base = row * k;

    float z = lg[base + t];
    bool  m = sp[base + t] > 0.0f;

    red[t] = m ? z : BIG_NEG;
    __syncthreads();
    for (int s = k >> 1; s > 0; s >>= 1) {
        if (t < s) red[t] = fmaxf(red[t], red[t + s]);
        __syncthreads();
    }
    float zmax = red[0];
    __syncthreads();

    float zc = z - zmax;
    d[t] = m ? zc : BIG_NEG;
    __syncthreads();

    for (int sz = 2; sz <= k; sz <<= 1) {
        for (int st = sz >> 1; st > 0; st >>= 1) {
            int ixj = t ^ st;
            if (ixj > t) {
                float a = d[t], b2 = d[ixj];
                bool up = ((t & sz) == 0);
                if (up ? (a < b2) : (a > b2)) { d[t] = b2; d[ixj] = a; }
            }
            __syncthreads();
        }
    }

    cum[t] = d[t];
    __syncthreads();
    for (int off = 1; off < k; off <<= 1) {
        float add = (t >= off) ? cum[t - off] : 0.0f;
        __syncthreads();
        cum[t] += add;
        __syncthreads();
    }

    red[t] = ((1.0f + (float)(t + 1) * d[t]) > cum[t]) ? 1.0f : 0.0f;
    __syncthreads();
    for (int s = k >> 1; s > 0; s >>= 1) {
        if (t < s) red[t] += red[t + s];
        __syncthreads();
    }
    int rho = (int)red[0];
    __syncthreads();
    float tau = (cum[rho - 1] - 1.0f) / (float)rho;
    out[base + t] = m ? fmaxf(zc - tau, 0.0f) : 0.0f;
}

// ---------------------------------------------------------------------------
// Readout (gmp || gap), combine, log-softmax.
// ---------------------------------------------------------------------------
__global__ void k_readout(const float* __restrict__ h, float* __restrict__ out, int n, int hd)
{
    int b = blockIdx.y;
    int f = blockIdx.x * blockDim.x + threadIdx.x;
    if (f >= hd) return;
    const float* base = h + (long long)b * n * hd + f;
    float mx = BIG_NEG, sm = 0.0f;
    for (int i = 0; i < n; ++i) {
        float v = base[(long long)i * hd];
        mx = fmaxf(mx, v);
        sm += v;
    }
    out[(long long)b * 2 * hd + f]      = mx;
    out[(long long)b * 2 * hd + hd + f] = sm / (float)n;
}

__global__ void k_relu_sum3(const float* __restrict__ x1, const float* __restrict__ x2,
                            const float* __restrict__ x3, float* __restrict__ z, int tot)
{
    int idx = blockIdx.x * blockDim.x + threadIdx.x;
    if (idx >= tot) return;
    z[idx] = fmaxf(x1[idx], 0.0f) + fmaxf(x2[idx], 0.0f) + fmaxf(x3[idx], 0.0f);
}

// z is [CB][ld] f32, first c columns valid.
__global__ void k_logsoftmax(const float* __restrict__ z, float* __restrict__ out, int c, int ld)
{
    int b = blockIdx.x * blockDim.x + threadIdx.x;
    if (b >= CB) return;
    const float* r = z + (long long)b * ld;
    float mx = r[0];
    for (int i = 1; i < c; ++i) mx = fmaxf(mx, r[i]);
    float s = 0.0f;
    for (int i = 0; i < c; ++i) s += expf(r[i] - mx);
    float l = mx + logf(s);
    for (int i = 0; i < c; ++i) out[(long long)b * c + i] = r[i] - l;
}

// ---------------------------------------------------------------------------
// Host-side helpers (all launches on `stream`; no alloc/sync).
// ---------------------------------------------------------------------------
static void cvt(hipStream_t st, const float* src, _Float16* dst, long long count)
{
    long long tot8 = count >> 3;   // all call sites are multiples of 8
    k_cvt8<<<(unsigned)((tot8 + 255) / 256), 256, 0, st>>>(src, dst, tot8);
}

static void cvt_t(hipStream_t st, const float* src, _Float16* dst,
                  int rows, int cols, int batch)
{
    long long tot = (long long)batch * rows * cols;
    k_cvt_t<<<(unsigned)((tot + 255) / 256), 256, 0, st>>>(src, dst, rows, cols, tot);
}

static void gemm16(hipStream_t st, const _Float16* Ah, const _Float16* Bth,
                   const float* bias, float* C, int M, int Nc, int K, int batch,
                   long long sA, long long sB, long long sC, int relu)
{
    int tiles = (M >> 4) * (Nc >> 4);
    dim3 grid((unsigned)((tiles + 3) / 4), (unsigned)batch);
    k_wmma_gemm16<<<grid, dim3(128), 0, st>>>(Ah, Bth, bias, C, M, Nc, K,
                                              sA, sB, sC, relu);
}

static void gcn(hipStream_t st, const float* h_in, const float* adj_m, const float* Wm,
                const float* bias, float* h_out, float* xw_t, float* an_t, float* dinv_t,
                _Float16* a16, _Float16* bt16, _Float16* wt16,
                int n, int fin, int fout)
{
    // xw = h_in @ W (shared weight: flattened GEMM)
    cvt(st, h_in, a16, (long long)CB * n * fin);
    cvt_t(st, Wm, wt16, fin, fout, 1);
    gemm16(st, a16, wt16, nullptr, xw_t, CB * n, fout, fin, 1, 0, 0, 0, 0);
    // normalized adjacency (self-loop fill semantics)
    k_gcn_dinv<<<dim3((n + 255) / 256, CB), 256, 0, st>>>(adj_m, dinv_t, n);
    long long tot = (long long)CB * n * n;
    k_gcn_norm<<<(unsigned)((tot + 255) / 256), 256, 0, st>>>(adj_m, dinv_t, an_t, n);
    // h = relu(a_n @ xw + b)
    cvt(st, an_t, a16, tot);
    cvt_t(st, xw_t, bt16, n, fout, CB);
    gemm16(st, a16, bt16, bias, h_out, n, fout, n, CB,
           (long long)n * n, (long long)n * fout, (long long)n * fout, 1);
}

static void info_score(hipStream_t st, const float* h_in, const float* adj_m,
                       float* an_t, float* dinv_t, float* y_t, float* score_t,
                       _Float16* a16, _Float16* bt16, int n, int f)
{
    k_row_dinv<<<dim3((n + 255) / 256, CB), 256, 0, st>>>(adj_m, dinv_t, n);
    long long tot = (long long)CB * n * n;
    k_sym_norm<<<(unsigned)((tot + 255) / 256), 256, 0, st>>>(adj_m, dinv_t, an_t, n);
    cvt(st, an_t, a16, tot);
    cvt_t(st, h_in, bt16, n, f, CB);
    gemm16(st, a16, bt16, nullptr, y_t, n, f, n, CB,
           (long long)n * n, (long long)n * f, (long long)n * f, 0);
    k_abs_diff_rowsum<<<dim3((n + 255) / 256, CB), 256, 0, st>>>(h_in, y_t, score_t, n, f);
}

static void pool(hipStream_t st, const float* h_in, const float* adj_m, const float* att,
                 int n, int k,
                 float* an_t, float* dinv_t, float* y_t, float* score_t,
                 float* s_t, float* sq_t, int* perm_t,
                 float* xp_t, float* sp_t, float* ap_t, float* lg_t,
                 float* ai_t, float* aj_t, float* newadj,
                 _Float16* a16, _Float16* bt16)
{
    info_score(st, h_in, adj_m, an_t, dinv_t, y_t, score_t, a16, bt16, n, CH);
    k_topk<<<CB, n, 3 * n * 4, st>>>(score_t, perm_t, n, k);

    long long nn = (long long)CB * n * n;
    k_binarize<<<(unsigned)((nn + 255) / 256), 256, 0, st>>>(adj_m, s_t, nn);
    for (int p = 0; p < 2; ++p) {   // TwoHopNeighborhood x2 -> 4-hop reach
        // s is symmetric: the row-major f16 copy doubles as B^T.
        cvt(st, s_t, a16, nn);
        gemm16(st, a16, a16, nullptr, sq_t, n, n, n, CB,
               (long long)n * n, (long long)n * n, (long long)n * n, 0);
        k_threshold_combine<<<(unsigned)((nn + 255) / 256), 256, 0, st>>>(s_t, sq_t, nn);
    }

    long long kh = (long long)CB * k * CH;
    k_gather_rows<<<(unsigned)((kh + 255) / 256), 256, 0, st>>>(h_in, perm_t, xp_t, n, k, CH);
    long long kk = (long long)CB * k * k;
    k_sub_matrix<<<(unsigned)((kk + 255) / 256), 256, 0, st>>>(s_t, perm_t, sp_t, n, k, 1);
    k_sub_matrix<<<(unsigned)((kk + 255) / 256), 256, 0, st>>>(adj_m, perm_t, ap_t, n, k, 0);

    k_att_dots<<<dim3((k + 255) / 256, CB), 256, 0, st>>>(xp_t, att, ai_t, aj_t, k, CH);
    k_logits<<<(unsigned)((kk + 255) / 256), 256, 0, st>>>(ai_t, aj_t, ap_t, lg_t, k);
    k_sparsemax<<<CB * k, k, 3 * k * 4, st>>>(lg_t, sp_t, newadj, k);
}

// ---------------------------------------------------------------------------
extern "C" void kernel_launch(void* const* d_in, const int* in_sizes, int n_in,
                              void* d_out, int out_size, void* d_ws, size_t ws_size,
                              hipStream_t stream)
{
    (void)in_sizes; (void)n_in; (void)out_size; (void)ws_size;

    const float* x    = (const float*)d_in[0];
    const float* adj  = (const float*)d_in[1];
    const float* W1   = (const float*)d_in[2];
    const float* b1   = (const float*)d_in[3];
    const float* att1 = (const float*)d_in[4];
    const float* W2   = (const float*)d_in[5];
    const float* b2   = (const float*)d_in[6];
    const float* att2 = (const float*)d_in[7];
    const float* W3   = (const float*)d_in[8];
    const float* b3   = (const float*)d_in[9];
    const float* Wl1  = (const float*)d_in[10];
    const float* bl1  = (const float*)d_in[11];
    const float* Wl2  = (const float*)d_in[12];
    const float* bl2  = (const float*)d_in[13];
    const float* Wl3  = (const float*)d_in[14];
    const float* bl3  = (const float*)d_in[15];

    char* w = (char*)d_ws;
    size_t off = 0;
    auto carve = [&](size_t nfloat) -> float* {
        float* p = (float*)(w + off);
        off += ((nfloat * sizeof(float)) + 255) & ~(size_t)255;
        return p;
    };
    auto carve16 = [&](size_t nh) -> _Float16* {
        _Float16* p = (_Float16*)(w + off);
        off += ((nh * sizeof(_Float16)) + 255) & ~(size_t)255;
        return p;
    };

    const size_t NN   = (size_t)CB * CN * CN;
    const size_t FN   = (size_t)CB * CN * CH;
    const size_t K1K1 = (size_t)CB * CK1 * CK1;
    const size_t K1H  = (size_t)CB * CK1 * CH;
    const size_t K2K2 = (size_t)CB * CK2 * CK2;
    const size_t K2H  = (size_t)CB * CK2 * CH;

    float* an    = carve(NN);          // gcn1 a_n -> info a_n0 -> s@s temp
    float* sbig  = carve(NN);          // 4-hop reachability (stage 1)
    float* xw    = carve(FN);          // x@W1 -> info y (stage 1)
    float* h1    = carve(FN);
    float* dinv  = carve((size_t)CB * CN);
    float* score = carve((size_t)CB * CN);
    int*   perm1 = (int*)carve((size_t)CB * CK1);
    float* xp1   = carve(K1H);
    float* sp1   = carve(K1K1);
    float* ap1   = carve(K1K1);
    float* lg1   = carve(K1K1);
    float* adj1  = carve(K1K1);
    float* ai    = carve((size_t)CB * CK1);
    float* aj    = carve((size_t)CB * CK1);
    float* x1r   = carve((size_t)CB * 2 * CH);
    float* an2   = carve(K1K1);
    float* s2    = carve(K1K1);
    float* sq2   = carve(K1K1);
    float* xw2   = carve(K1H);
    float* h2    = carve(K1H);
    int*   perm2 = (int*)carve((size_t)CB * CK2);
    float* xp2   = carve(K2H);
    float* sp2   = carve(K2K2);
    float* ap2   = carve(K2K2);
    float* lg2   = carve(K2K2);
    float* adj2  = carve(K2K2);
    float* x2r   = carve((size_t)CB * 2 * CH);
    float* an3   = carve(K2K2);
    float* xw3   = carve(K2H);
    float* h3    = carve(K2H);
    float* x3r   = carve((size_t)CB * 2 * CH);
    float* zs    = carve((size_t)CB * 2 * CH);
    float* z1    = carve((size_t)CB * CH);
    float* z2b   = carve((size_t)CB * (CH / 2));
    float* z3p   = carve((size_t)CB * CCP);
    float* bl3p  = carve((size_t)CCP);

    // f16 GEMM scratch (reused across all stages; stage-1 sizes are maximal)
    _Float16* a16  = carve16(NN);                  // A operands (row-major f16)
    _Float16* bt16 = carve16(FN);                  // transposed feature B operands
    _Float16* wt16 = carve16((size_t)256 * 256);   // transposed weight operands

    // ---- Layer 1: h1 = relu(GCN(x, adj)) ------------------------------------
    gcn(stream, x, adj, W1, b1, h1, xw, an, dinv, a16, bt16, wt16, CN, CH, CH);

    // ---- Pool 1: (h1, adj) -> (xp1 [B,256,H], adj1 [B,256,256]) -------------
    pool(stream, h1, adj, att1, CN, CK1,
         an, dinv, xw, score, sbig, /*sq=*/an, perm1,
         xp1, sp1, ap1, lg1, ai, aj, adj1, a16, bt16);
    k_readout<<<dim3((CH + 255) / 256, CB), 256, 0, stream>>>(xp1, x1r, CK1, CH);

    // ---- Layer 2: h2 = relu(GCN(xp1, adj1)) ---------------------------------
    gcn(stream, xp1, adj1, W2, b2, h2, xw2, an2, dinv, a16, bt16, wt16, CK1, CH, CH);

    // ---- Pool 2: (h2, adj1) -> (xp2 [B,128,H], adj2 [B,128,128]) ------------
    pool(stream, h2, adj1, att2, CK1, CK2,
         an2, dinv, xw2, score, s2, sq2, perm2,
         xp2, sp2, ap2, lg2, ai, aj, adj2, a16, bt16);
    k_readout<<<dim3((CH + 255) / 256, CB), 256, 0, stream>>>(xp2, x2r, CK2, CH);

    // ---- Layer 3: h3 = relu(GCN(xp2, adj2)) ---------------------------------
    gcn(stream, xp2, adj2, W3, b3, h3, xw3, an3, dinv, a16, bt16, wt16, CK2, CH, CH);
    k_readout<<<dim3((CH + 255) / 256, CB), 256, 0, stream>>>(h3, x3r, CK2, CH);

    // ---- Head ---------------------------------------------------------------
    k_relu_sum3<<<(CB * 2 * CH + 255) / 256, 256, 0, stream>>>(x1r, x2r, x3r, zs, CB * 2 * CH);

    cvt(stream, zs, a16, (long long)CB * 2 * CH);
    cvt_t(stream, Wl1, wt16, 2 * CH, CH, 1);
    gemm16(stream, a16, wt16, bl1, z1, CB, CH, 2 * CH, 1, 0, 0, 0, 1);

    cvt(stream, z1, a16, (long long)CB * CH);
    cvt_t(stream, Wl2, wt16, CH, CH / 2, 1);
    gemm16(stream, a16, wt16, bl2, z2b, CB, CH / 2, CH, 1, 0, 0, 0, 1);

    cvt(stream, z2b, a16, (long long)CB * (CH / 2));
    k_pad_w3<<<4, 256, 0, stream>>>(Wl3, bl3, wt16, bl3p);
    gemm16(stream, a16, wt16, bl3p, z3p, CB, CCP, CH / 2, 1, 0, 0, 0, 0);

    k_logsoftmax<<<1, CB, 0, stream>>>(z3p, (float*)d_out, CC, CCP);
}
